// NLB_4741643895350
// MI455X (gfx1250) — matrix operand order repaired and checked by reading, not compiled, not run on hardware
//
#include <hip/hip_runtime.h>
#include <hip/hip_bf16.h>

#define N_SP 6400   // H*W
#define CCH  256    // channels
#define CI_  128    // inner channels
#define BATCH 2

typedef __attribute__((ext_vector_type(16))) __bf16 v16bf;
typedef __attribute__((ext_vector_type(8)))  float  v8f;

union FragBF {
    v16bf v;
    unsigned short u[16];
    uint4 q[2];
};

__device__ __forceinline__ unsigned short f2bfu(float f) {
    unsigned int u = __float_as_uint(f);
    unsigned int r = u + 0x7FFFu + ((u >> 16) & 1u);   // round-to-nearest-even
    return (unsigned short)(r >> 16);
}

__device__ __forceinline__ v8f zero8() {
    v8f z;
#pragma unroll
    for (int i = 0; i < 8; ++i) z[i] = 0.0f;
    return z;
}

__device__ __forceinline__ v8f wmma_bf(FragBF a, FragBF b, v8f c) {
    // D = A(16x32 bf16) * B(32x16 bf16) + C(16x16 f32)
    return __builtin_amdgcn_wmma_f32_16x16x32_bf16(
        false, a.v, false, b.v, (short)0, c, false, false);
}

// ---------------------------------------------------------------------------
// K0: fp32 -> bf16 conversion (weights)
// ---------------------------------------------------------------------------
__global__ void k_cvt_bf16(const float* __restrict__ src,
                           unsigned short* __restrict__ dst, int n) {
    int i = blockIdx.x * blockDim.x + threadIdx.x;
    if (i < n) dst[i] = f2bfu(src[i]);
}

// ---------------------------------------------------------------------------
// K1: projections theta/phi/g.  proj[p, ci] = sum_c x[c, p] * w[ci, c] + b[ci]
// Block: 768 threads = 24 waves: wave = proj(3) x ci-tile(8).
// x tile converted to bf16 and TRANSPOSED during staging into A-fragment
// layout xsT[p][c] (pitch 264), so A frags are two ds_load_b128 per k-step.
// theta/phi stored row-major [N, Ci] bf16; g stored TRANSPOSED [Ci, N] bf16.
// ---------------------------------------------------------------------------
__global__ __launch_bounds__(768) void k_proj(
    const float* __restrict__ x,
    const unsigned short* __restrict__ wT, const unsigned short* __restrict__ wP,
    const unsigned short* __restrict__ wG,
    const float* __restrict__ bT, const float* __restrict__ bP,
    const float* __restrict__ bG,
    unsigned short* __restrict__ theta, unsigned short* __restrict__ phi,
    unsigned short* __restrict__ gT)
{
    __shared__ __align__(16) unsigned short xsT[16 * 264];  // [p][c] bf16
    const int b  = blockIdx.y;
    const int p0 = blockIdx.x * 16;
    const int tid = threadIdx.x;

    for (int i = tid; i < CCH * 16; i += 768) {
        int c = i >> 4, p = i & 15;
        xsT[p * 264 + c] = f2bfu(x[((size_t)b * CCH + c) * N_SP + p0 + p]);
    }
    __syncthreads();

    const int wave = tid >> 5, lane = tid & 31;
    const int g = lane >> 4, pl = lane & 15;
    const int proj = wave >> 3, ci0 = (wave & 7) * 16;

    const unsigned short* wptr;
    const float* bias;
    if (proj == 0)      { wptr = wT; bias = bT; }
    else if (proj == 1) { wptr = wP; bias = bP; }
    else                { wptr = wG; bias = bG; }

    v8f acc = zero8();
#pragma unroll
    for (int k0 = 0; k0 < CCH; k0 += 32) {
        FragBF a;
        const unsigned short* pa = &xsT[pl * 264 + k0 + 8 * g];
        a.q[0] = *reinterpret_cast<const uint4*>(pa);
        a.q[1] = *reinterpret_cast<const uint4*>(pa + 16);
        FragBF bb;
        const unsigned short* wp = wptr + (size_t)(ci0 + pl) * CCH + k0 + 8 * g;
        bb.q[0] = *reinterpret_cast<const uint4*>(wp);
        bb.q[1] = *reinterpret_cast<const uint4*>(wp + 16);
        acc = wmma_bf(a, bb, acc);
    }

    float bv = bias[ci0 + pl];
    if (proj < 2) {
        unsigned short* outp = ((proj == 0) ? theta : phi) + (size_t)b * N_SP * CI_;
#pragma unroll
        for (int r = 0; r < 8; ++r)
            outp[(size_t)(p0 + r + 8 * g) * CI_ + ci0 + pl] = f2bfu(acc[r] + bv);
    } else {
        union { uint4 q; unsigned short u[8]; } pk;
#pragma unroll
        for (int r = 0; r < 8; ++r) pk.u[r] = f2bfu(acc[r] + bv);
        *reinterpret_cast<uint4*>(gT + (size_t)b * CI_ * N_SP +
                                  (size_t)(ci0 + pl) * N_SP + p0 + 8 * g) = pk.q;
    }
}

// ---------------------------------------------------------------------------
// K2: column softmax stats. f[n,m] = theta[n,:].phi[m,:]; softmax over n.
// Block: 256 thr = 8 waves, each wave owns a 16-column strip (m), loops all n.
// phi B-frags hoisted in registers; theta tile staged in LDS per n-step.
// Online (max, sum) per lane; merge lane<->lane+16 with shfl_xor.
// ---------------------------------------------------------------------------
__global__ __launch_bounds__(256) void k_colstats(
    const unsigned short* __restrict__ theta, const unsigned short* __restrict__ phi,
    float* __restrict__ colmax, float* __restrict__ colsum)
{
    __shared__ __align__(16) unsigned short ts[16 * 136];
    const int b = blockIdx.y;
    const int tid = threadIdx.x, wave = tid >> 5, lane = tid & 31;
    const int g = lane >> 4, pl = lane & 15;
    const int m0 = blockIdx.x * 128 + wave * 16;
    const unsigned short* th = theta + (size_t)b * N_SP * CI_;
    const unsigned short* ph = phi   + (size_t)b * N_SP * CI_;

    FragBF bfr[4];
#pragma unroll
    for (int kk = 0; kk < 4; ++kk) {
        const unsigned short* p = ph + (size_t)(m0 + pl) * CI_ + kk * 32 + 8 * g;
        bfr[kk].q[0] = *reinterpret_cast<const uint4*>(p);
        bfr[kk].q[1] = *reinterpret_cast<const uint4*>(p + 16);
    }

    float mrun = -3.0e38f, srun = 0.0f;
    for (int n0 = 0; n0 < N_SP; n0 += 16) {
        __syncthreads();
        {   // stage theta[n0..n0+15][0..127] : 256 uint4, one per thread
            int row = tid >> 4, col = (tid & 15) * 8;
            *reinterpret_cast<uint4*>(&ts[row * 136 + col]) =
                *reinterpret_cast<const uint4*>(&th[(size_t)(n0 + row) * CI_ + col]);
        }
        __syncthreads();

        v8f facc = zero8();
#pragma unroll
        for (int kk = 0; kk < 4; ++kk) {
            FragBF a;
            const unsigned short* p = &ts[pl * 136 + kk * 32 + 8 * g];
            a.q[0] = *reinterpret_cast<const uint4*>(p);
            a.q[1] = *reinterpret_cast<const uint4*>(p + 16);
            facc = wmma_bf(a, bfr[kk], facc);
        }
        float tmax = facc[0];
#pragma unroll
        for (int r = 1; r < 8; ++r) tmax = fmaxf(tmax, facc[r]);
        float nm = fmaxf(mrun, tmax);
        float s = 0.0f;
#pragma unroll
        for (int r = 0; r < 8; ++r) s += __expf(facc[r] - nm);
        srun = srun * __expf(mrun - nm) + s;
        mrun = nm;
    }

    float mo = __shfl_xor(mrun, 16, 32);
    float so = __shfl_xor(srun, 16, 32);
    float mt = fmaxf(mrun, mo);
    float st = srun * __expf(mrun - mt) + so * __expf(mo - mt);
    if (lane < 16) {
        colmax[(size_t)b * N_SP + m0 + lane] = mt;
        colsum[(size_t)b * N_SP + m0 + lane] = st;
    }
}

// ---------------------------------------------------------------------------
// K3: y[n,:] = sum_m exp(f[n,m]-colmax[m])/colsum[m] * g[m,:]
// Block: 256 thr = 8 waves; wave owns 16 n-rows (A-frags hoisted in regs).
// Per 32-m chunk: stage phi (B layout) + gT (B layout) in LDS shared by
// all waves; compute f tiles, exponentiate, transpose p through per-wave
// LDS tile into A layout, then 8 WMMAs accumulate all 128 ci.
// ---------------------------------------------------------------------------
__global__ __launch_bounds__(256) void k_attn(
    const unsigned short* __restrict__ theta, const unsigned short* __restrict__ phi,
    const unsigned short* __restrict__ gT,
    const float* __restrict__ colmax, const float* __restrict__ colsum,
    unsigned short* __restrict__ ybf)
{
    __shared__ __align__(16) unsigned short ps[32 * 136];   // phi  [m][k]
    __shared__ __align__(16) unsigned short gs[128 * 40];   // gT   [ci][m]
    __shared__ __align__(16) unsigned short pT[8][16 * 40]; // per-wave p [n][k]
    const int b = blockIdx.y;
    const int tid = threadIdx.x, wave = tid >> 5, lane = tid & 31;
    const int g = lane >> 4, pl = lane & 15;
    const int p0 = blockIdx.x * 128 + wave * 16;
    const unsigned short* th  = theta + (size_t)b * N_SP * CI_;
    const unsigned short* ph  = phi   + (size_t)b * N_SP * CI_;
    const unsigned short* gTb = gT    + (size_t)b * CI_ * N_SP;
    const float* cmb = colmax + (size_t)b * N_SP;
    const float* csb = colsum + (size_t)b * N_SP;

    FragBF afr[4];
#pragma unroll
    for (int kk = 0; kk < 4; ++kk) {
        const unsigned short* p = th + (size_t)(p0 + pl) * CI_ + kk * 32 + 8 * g;
        afr[kk].q[0] = *reinterpret_cast<const uint4*>(p);
        afr[kk].q[1] = *reinterpret_cast<const uint4*>(p + 16);
    }
    v8f yacc[8];
#pragma unroll
    for (int t = 0; t < 8; ++t) yacc[t] = zero8();
    unsigned short* myp = pT[wave];

    for (int m0 = 0; m0 < N_SP; m0 += 32) {
        __syncthreads();
#pragma unroll
        for (int i = 0; i < 2; ++i) {   // stage phi[m0..m0+31][0..127]
            int t = tid + i * 256;
            int row = t >> 4, col = (t & 15) * 8;
            *reinterpret_cast<uint4*>(&ps[row * 136 + col]) =
                *reinterpret_cast<const uint4*>(&ph[(size_t)(m0 + row) * CI_ + col]);
        }
#pragma unroll
        for (int i = 0; i < 2; ++i) {   // stage gT[0..127][m0..m0+31]
            int t = tid + i * 256;
            int ci = t >> 2, part = (t & 3) * 8;
            *reinterpret_cast<uint4*>(&gs[ci * 40 + part]) =
                *reinterpret_cast<const uint4*>(&gTb[(size_t)ci * N_SP + m0 + part]);
        }
        // prefetch next chunk into caches while we compute this one
        if (m0 + 32 < N_SP) {
            __builtin_prefetch(&ph[(size_t)(m0 + 32 + (tid & 31)) * CI_], 0, 3);
            if (tid < 128)
                __builtin_prefetch(&gTb[(size_t)tid * N_SP + m0 + 32], 0, 3);
        }
        __syncthreads();

#pragma unroll
        for (int s = 0; s < 2; ++s) {
            v8f facc = zero8();
#pragma unroll
            for (int kk = 0; kk < 4; ++kk) {
                FragBF bb;
                const unsigned short* p = &ps[(s * 16 + pl) * 136 + kk * 32 + 8 * g];
                bb.q[0] = *reinterpret_cast<const uint4*>(p);
                bb.q[1] = *reinterpret_cast<const uint4*>(p + 16);
                facc = wmma_bf(afr[kk], bb, facc);
            }
            int m = m0 + s * 16 + pl;
            float cm = cmb[m];
            float rs = 1.0f / csb[m];
#pragma unroll
            for (int r = 0; r < 8; ++r)
                myp[(r + 8 * g) * 40 + s * 16 + pl] = f2bfu(__expf(facc[r] - cm) * rs);
        }
        __syncthreads();   // make p-tile writes visible (also fences stages)

        FragBF a2;
        {
            const unsigned short* p = &myp[pl * 40 + 8 * g];
            a2.q[0] = *reinterpret_cast<const uint4*>(p);
            a2.q[1] = *reinterpret_cast<const uint4*>(p + 16);
        }
#pragma unroll
        for (int t = 0; t < 8; ++t) {
            FragBF b2;
            const unsigned short* p = &gs[(t * 16 + pl) * 40 + 8 * g];
            b2.q[0] = *reinterpret_cast<const uint4*>(p);
            b2.q[1] = *reinterpret_cast<const uint4*>(p + 16);
            yacc[t] = wmma_bf(a2, b2, yacc[t]);
        }
    }

    unsigned short* yb = ybf + (size_t)b * N_SP * CI_;
#pragma unroll
    for (int t = 0; t < 8; ++t)
#pragma unroll
        for (int r = 0; r < 8; ++r)
            yb[(size_t)(p0 + r + 8 * g) * CI_ + t * 16 + pl] = f2bfu(yacc[t][r]);
}

// ---------------------------------------------------------------------------
// K4: out[b,co,p] = W_b[co] + sum_ci W_w[co,ci]*y[p,ci] + x[b,co,p]
// Block: 256 thr = 8 waves; wave = one 16-co tile; grid.y picks co half.
// ---------------------------------------------------------------------------
__global__ __launch_bounds__(256) void k_outproj(
    const unsigned short* __restrict__ ybf, const unsigned short* __restrict__ wW,
    const float* __restrict__ Wb, const float* __restrict__ x,
    float* __restrict__ out)
{
    const int b = blockIdx.z;
    const int p0 = blockIdx.x * 16;
    const int tid = threadIdx.x, wave = tid >> 5, lane = tid & 31;
    const int g = lane >> 4, pl = lane & 15;
    const int co0 = blockIdx.y * 128 + wave * 16;
    const unsigned short* yb = ybf + (size_t)b * N_SP * CI_;

    v8f acc = zero8();
#pragma unroll
    for (int kk = 0; kk < 4; ++kk) {
        FragBF a;
        const unsigned short* pa = yb + (size_t)(p0 + pl) * CI_ + kk * 32 + 8 * g;
        a.q[0] = *reinterpret_cast<const uint4*>(pa);
        a.q[1] = *reinterpret_cast<const uint4*>(pa + 16);
        FragBF bb;
        const unsigned short* pb = wW + (size_t)(co0 + pl) * CI_ + kk * 32 + 8 * g;
        bb.q[0] = *reinterpret_cast<const uint4*>(pb);
        bb.q[1] = *reinterpret_cast<const uint4*>(pb + 16);
        acc = wmma_bf(a, bb, acc);
    }

    int co = co0 + pl;
    float bv = Wb[co];
    const float* xr = x + ((size_t)b * CCH + co) * N_SP + p0 + 8 * g;
    float4 x0 = *reinterpret_cast<const float4*>(xr);
    float4 x1 = *reinterpret_cast<const float4*>(xr + 4);
    float* orow = out + ((size_t)b * CCH + co) * N_SP + p0 + 8 * g;
    float4 o0, o1;
    o0.x = acc[0] + bv + x0.x; o0.y = acc[1] + bv + x0.y;
    o0.z = acc[2] + bv + x0.z; o0.w = acc[3] + bv + x0.w;
    o1.x = acc[4] + bv + x1.x; o1.y = acc[5] + bv + x1.y;
    o1.z = acc[6] + bv + x1.z; o1.w = acc[7] + bv + x1.w;
    *reinterpret_cast<float4*>(orow) = o0;
    *reinterpret_cast<float4*>(orow + 4) = o1;
}

// ---------------------------------------------------------------------------
extern "C" void kernel_launch(void* const* d_in, const int* in_sizes, int n_in,
                              void* d_out, int out_size, void* d_ws, size_t ws_size,
                              hipStream_t stream) {
    const float* x    = (const float*)d_in[0];
    const float* g_w  = (const float*)d_in[1];
    const float* g_b  = (const float*)d_in[2];
    const float* th_w = (const float*)d_in[3];
    const float* th_b = (const float*)d_in[4];
    const float* ph_w = (const float*)d_in[5];
    const float* ph_b = (const float*)d_in[6];
    const float* W_w  = (const float*)d_in[7];
    const float* W_b  = (const float*)d_in[8];
    float* out = (float*)d_out;

    char* ws = (char*)d_ws;
    size_t off = 0;
    auto alloc = [&](size_t bytes) -> char* {
        char* p = ws + off;
        off += (bytes + 255) & ~(size_t)255;
        return p;
    };

    unsigned short* wT    = (unsigned short*)alloc((size_t)CI_ * CCH * 2);
    unsigned short* wP    = (unsigned short*)alloc((size_t)CI_ * CCH * 2);
    unsigned short* wG    = (unsigned short*)alloc((size_t)CI_ * CCH * 2);
    unsigned short* wWb16 = (unsigned short*)alloc((size_t)CCH * CI_ * 2);
    unsigned short* theta = (unsigned short*)alloc((size_t)BATCH * N_SP * CI_ * 2);
    unsigned short* phi   = (unsigned short*)alloc((size_t)BATCH * N_SP * CI_ * 2);
    unsigned short* gT    = (unsigned short*)alloc((size_t)BATCH * CI_ * N_SP * 2);
    unsigned short* ybf   = (unsigned short*)alloc((size_t)BATCH * N_SP * CI_ * 2);
    float* colmax = (float*)alloc((size_t)BATCH * N_SP * 4);
    float* colsum = (float*)alloc((size_t)BATCH * N_SP * 4);

    const int NW = CI_ * CCH;   // 32768
    k_cvt_bf16<<<(NW + 255) / 256, 256, 0, stream>>>(th_w, wT, NW);
    k_cvt_bf16<<<(NW + 255) / 256, 256, 0, stream>>>(ph_w, wP, NW);
    k_cvt_bf16<<<(NW + 255) / 256, 256, 0, stream>>>(g_w, wG, NW);
    k_cvt_bf16<<<(NW + 255) / 256, 256, 0, stream>>>(W_w, wWb16, NW);

    k_proj<<<dim3(N_SP / 16, BATCH), 768, 0, stream>>>(
        x, wT, wP, wG, th_b, ph_b, g_b, theta, phi, gT);
    k_colstats<<<dim3(N_SP / 128, BATCH), 256, 0, stream>>>(
        theta, phi, colmax, colsum);
    k_attn<<<dim3(N_SP / 128, BATCH), 256, 0, stream>>>(
        theta, phi, gT, colmax, colsum, ybf);
    k_outproj<<<dim3(N_SP / 16, 2, BATCH), 256, 0, stream>>>(
        ybf, wWb16, W_b, x, out);
}
